// MTGRU_48722109006521
// MI455X (gfx1250) — matrix operand order repaired and checked by reading, not compiled, not run on hardware
//
#include <hip/hip_runtime.h>

// ---------------- CDNA5 WMMA types ----------------
typedef __attribute__((ext_vector_type(16))) _Float16 v16h;
typedef __attribute__((ext_vector_type(8)))  float    v8f;

#define D_IN    768
#define HID     256
#define G3      768                 // 3*HID
#define BATCH   64
#define SEQ     512
#define ROWS    (BATCH*SEQ)         // 32768
#define KT_IN   (D_IN/32)           // 24 k-tiles for the big GEMM
#define KT_H    (HID/32)            // 8 k-tiles for recurrent GEMMs
#define NT_G3   (G3/16)             // 48 n-tiles (3 gates x 16)
#define NWG_REC 8
#define THR_REC 256
#define LDSW_HALFS (3*2*3*KT_H*512) // 73728 halfs = 144KB staged weights

union AF { v16h h; unsigned u[8]; };

__device__ __forceinline__ v8f wmma_f16(v16h a, v16h b, v8f c) {
  return __builtin_amdgcn_wmma_f32_16x16x32_f16(false, a, false, b, (short)0, c, false, false);
}

// B fragment from pre-swizzled global storage [ntile][ktile][lane][16 halfs]
__device__ __forceinline__ v16h load_bfrag(const _Float16* __restrict__ wsw,
                                           int ntile, int kt, int KT, int lane) {
  const _Float16* p = wsw + ((((size_t)ntile * KT + kt) << 9)) + (lane << 4);
  return *(const v16h*)p;
}

// B fragment from LDS stage: [mat][ntl][gate][kt][lane][16]
__device__ __forceinline__ v16h lds_bfrag(const _Float16* ldsW,
                                          int mat, int ntl, int gate, int kt, int lane) {
  const _Float16* p = ldsW + ((((((mat * 2 + ntl) * 3 + gate) << 3) + kt) << 9)) + (lane << 4);
  return *(const v16h*)p;
}

// A fragment from row-major f16 [rows][ldk]: lane m=lane&15, half h=lane>>4
__device__ __forceinline__ AF load_afrag(const _Float16* __restrict__ base,
                                         int row, int ldk, int k0, int lane) {
  AF a;
  const unsigned* p = (const unsigned*)(base + (size_t)row * ldk + k0);
  const int b0 = (lane >> 4) << 2;
#pragma unroll
  for (int d = 0; d < 4; ++d) { a.u[d] = p[b0 + d]; a.u[4 + d] = p[8 + b0 + d]; }
  return a;
}

__device__ __forceinline__ float sigm(float x) { return 1.0f / (1.0f + __expf(-x)); }

// ---------------- prep kernels ----------------
__global__ void k_cvt_x(const float* __restrict__ X, _Float16* __restrict__ Xh, int n4) {
  int i = blockIdx.x * blockDim.x + threadIdx.x;
  if (i >= n4) return;
  float4 f = ((const float4*)X)[i];
  union { _Float16 h[4]; unsigned long long u; } o;
  o.h[0] = (_Float16)f.x; o.h[1] = (_Float16)f.y;
  o.h[2] = (_Float16)f.z; o.h[3] = (_Float16)f.w;
  ((unsigned long long*)Xh)[i] = o.u;
}

// pack row-major f32 weight [K][768] into B-fragment layout (f16)
__global__ void k_pack_w(const float* __restrict__ src, int K, _Float16* __restrict__ dst) {
  const int KT = K >> 5;
  const int total = NT_G3 * KT * 512;
  int tid = blockIdx.x * blockDim.x + threadIdx.x;
  if (tid >= total) return;
  int j     = tid & 15;
  int lane  = (tid >> 4) & 31;
  int tile  = tid >> 9;
  int kt    = tile % KT;
  int ntile = tile / KT;
  int n = ntile * 16 + (lane & 15);
  int k = kt * 32 + ((j >> 3) << 4) + ((lane >> 4) << 3) + (j & 7);
  dst[tid] = (_Float16)src[(size_t)k * G3 + n];
}

__global__ void k_init(_Float16* h1f, _Float16* h2f, unsigned* syncv) {
  int i = blockIdx.x * blockDim.x + threadIdx.x;
  if (i < (2 * BATCH * HID / 2)) { ((unsigned*)h1f)[i] = 0u; ((unsigned*)h2f)[i] = 0u; }
  if (i < 2) syncv[i] = 0u;
}

// ---------------- big input-projection GEMM ----------------
// One A fragment feeds both W_ih1 and W_ih2x tiles: 8 WMMA per A load.
__global__ void __launch_bounds__(256)
k_gemm_in(const _Float16* __restrict__ Xh,
          const _Float16* __restrict__ Wsw1, const _Float16* __restrict__ Wsw2,
          const float* __restrict__ bias1,   const float* __restrict__ bias2,
          float* __restrict__ GI1, float* __restrict__ GI2) {
  const int lane = threadIdx.x & 31;
  const int wave = (blockIdx.x * blockDim.x + threadIdx.x) >> 5;  // 0..24575
  const int mt  = wave / 12;      // 2048 M-tiles
  const int ng  = wave % 12;      // 12 groups of 4 N-tiles

  v8f acc1[4] = {{}, {}, {}, {}};
  v8f acc2[4] = {{}, {}, {}, {}};
  const int row = mt * 16 + (lane & 15);
  for (int kt = 0; kt < KT_IN; ++kt) {
    AF a = load_afrag(Xh, row, D_IN, kt * 32, lane);
#pragma unroll
    for (int j = 0; j < 4; ++j) {
      acc1[j] = wmma_f16(a.h, load_bfrag(Wsw1, ng * 4 + j, kt, KT_IN, lane), acc1[j]);
      acc2[j] = wmma_f16(a.h, load_bfrag(Wsw2, ng * 4 + j, kt, KT_IN, lane), acc2[j]);
    }
  }
  const int hh = lane >> 4;
#pragma unroll
  for (int j = 0; j < 4; ++j) {
    int col = (ng * 4 + j) * 16 + (lane & 15);
    float b1 = bias1[col], b2 = bias2[col];
#pragma unroll
    for (int v = 0; v < 8; ++v) {
      int m = mt * 16 + v + hh * 8;
      GI1[(size_t)m * G3 + col] = acc1[j][v] + b1;
      GI2[(size_t)m * G3 + col] = acc2[j][v] + b2;
    }
  }
}

// ---------------- persistent recurrent kernel ----------------
__device__ __forceinline__ void grid_sync(unsigned* __restrict__ syncv, unsigned* localGen) {
  __syncthreads();
  if (threadIdx.x == 0) {
    __threadfence();
    unsigned target = ++(*localGen);
    if (atomicAdd(&syncv[0], 1u) == NWG_REC - 1u) {
      atomicExch(&syncv[0], 0u);
      __threadfence();
      atomicAdd(&syncv[1], 1u);      // release generation
    } else {
      while (atomicAdd(&syncv[1], 0u) < target) { __builtin_amdgcn_s_sleep(1); }
      __threadfence();
    }
  }
  __syncthreads();
}

__global__ void __launch_bounds__(THR_REC)
k_recurrent(const float* __restrict__ GI1, const float* __restrict__ GI2,
            const _Float16* __restrict__ Whh1s, const _Float16* __restrict__ Wih2hs,
            const _Float16* __restrict__ Whh2s,
            const float* __restrict__ b_hh1, const float* __restrict__ b_hh2,
            const float* __restrict__ tau1,  const float* __restrict__ tau2,
            _Float16* __restrict__ h1f, _Float16* __restrict__ h2f,
            float* __restrict__ out, unsigned* __restrict__ syncv) {
  extern __shared__ _Float16 ldsW[];   // [mat3][ntl2][gate3][kt8][512]
  const int tid  = threadIdx.x;
  const int lane = tid & 31;
  const int warp = tid >> 5;                 // 0..7
  const int mt   = warp >> 1;                // 4 M-tiles of the batch
  const int ntl  = warp & 1;                 // local N-tile
  const int nt   = blockIdx.x * 2 + ntl;     // global N-tile of H (0..15)
  const int hh   = lane >> 4;
  const int ncol = nt * 16 + (lane & 15);
  const int arow = mt * 16 + (lane & 15);

  // ---- stage this WG's weight slice (144 KB) into LDS once ----
  for (int f = 0; f < 144; ++f) {            // 3 mats x 2 ntl x 3 gates x 8 kt
    int mat = f / 48, r = f % 48;
    int sntl = r / 24, r2 = r % 24;
    int gate = r2 / 8, kt = r2 % 8;
    const _Float16* gsrc = (mat == 0) ? Whh1s : (mat == 1) ? Wih2hs : Whh2s;
    int ntile = gate * 16 + blockIdx.x * 2 + sntl;
    const unsigned* gp = (const unsigned*)(gsrc + (((size_t)ntile * KT_H + kt) << 9));
    ((unsigned*)&ldsW[(size_t)f << 9])[tid] = gp[tid];   // 256 dwords per fragment
  }
  __syncthreads();

  const float inv1 = 1.0f / tau1[0];
  const float inv2 = 1.0f / tau2[0];
  const float br1 = b_hh1[ncol], bz1 = b_hh1[HID + ncol], bn1 = b_hh1[2 * HID + ncol];
  const float br2 = b_hh2[ncol], bz2 = b_hh2[HID + ncol], bn2 = b_hh2[2 * HID + ncol];
  float h1prev[8], h2prev[8];
#pragma unroll
  for (int v = 0; v < 8; ++v) { h1prev[v] = 0.0f; h2prev[v] = 0.0f; }
  unsigned localGen = 0;

  for (int t = 0; t < SEQ; ++t) {
    const int cur = t & 1, prv = (t ^ 1) & 1;
    const _Float16* h1p = h1f + prv * (BATCH * HID);
    _Float16*       h1c = h1f + cur * (BATCH * HID);
    const _Float16* h2p = h2f + prv * (BATCH * HID);
    _Float16*       h2c = h2f + cur * (BATCH * HID);

    // ---- phase 1: h1 = GRU1(x_t, h1) with leak tau1 ----
    {
      v8f ar = {}, az = {}, an = {};
#pragma unroll
      for (int kt = 0; kt < KT_H; ++kt) {
        AF a = load_afrag(h1p, arow, HID, kt * 32, lane);
        ar = wmma_f16(a.h, lds_bfrag(ldsW, 0, ntl, 0, kt, lane), ar);
        az = wmma_f16(a.h, lds_bfrag(ldsW, 0, ntl, 1, kt, lane), az);
        an = wmma_f16(a.h, lds_bfrag(ldsW, 0, ntl, 2, kt, lane), an);
      }
#pragma unroll
      for (int v = 0; v < 8; ++v) {
        int m = mt * 16 + v + hh * 8;
        size_t gib = ((size_t)m * SEQ + t) * G3;
        float gir = GI1[gib + ncol];
        float giz = GI1[gib + HID + ncol];
        float gin = GI1[gib + 2 * HID + ncol];
        float hp = h1prev[v];
        float r  = sigm(gir + ar[v] + br1);
        float z  = sigm(giz + az[v] + bz1);
        float nn = tanhf(gin + r * (an[v] + bn1));
        float hnew = (1.0f - z) * nn + z * hp;
        float hout = (1.0f - inv1) * hp + inv1 * hnew;
        h1prev[v] = hout;
        out[((size_t)m * SEQ + t) * (2 * HID) + ncol] = hout;
        h1c[m * HID + ncol] = (_Float16)hout;
      }
      if (t + 1 < SEQ) {  // gfx1250 global_prefetch_b8: next step's GI rows
        __builtin_prefetch((const char*)(GI1 + ((size_t)arow * SEQ + (t + 1)) * G3 + ncol), 0, 1);
        __builtin_prefetch((const char*)(GI2 + ((size_t)arow * SEQ + (t + 1)) * G3 + ncol), 0, 1);
      }
    }
    grid_sync(syncv, &localGen);

    // ---- phase 2: h2 = GRU2([x_t, h1], h2) with leak tau2 ----
    {
      v8f ir = {}, iz = {}, in_ = {}, hr = {}, hz = {}, hn = {};
#pragma unroll
      for (int kt = 0; kt < KT_H; ++kt) {
        AF a1 = load_afrag(h1c, arow, HID, kt * 32, lane);
        AF a2 = load_afrag(h2p, arow, HID, kt * 32, lane);
        ir  = wmma_f16(a1.h, lds_bfrag(ldsW, 1, ntl, 0, kt, lane), ir);
        iz  = wmma_f16(a1.h, lds_bfrag(ldsW, 1, ntl, 1, kt, lane), iz);
        in_ = wmma_f16(a1.h, lds_bfrag(ldsW, 1, ntl, 2, kt, lane), in_);
        hr  = wmma_f16(a2.h, lds_bfrag(ldsW, 2, ntl, 0, kt, lane), hr);
        hz  = wmma_f16(a2.h, lds_bfrag(ldsW, 2, ntl, 1, kt, lane), hz);
        hn  = wmma_f16(a2.h, lds_bfrag(ldsW, 2, ntl, 2, kt, lane), hn);
      }
#pragma unroll
      for (int v = 0; v < 8; ++v) {
        int m = mt * 16 + v + hh * 8;
        size_t gib = ((size_t)m * SEQ + t) * G3;
        float gir = GI2[gib + ncol]           + ir[v];
        float giz = GI2[gib + HID + ncol]     + iz[v];
        float gin = GI2[gib + 2 * HID + ncol] + in_[v];
        float hp = h2prev[v];
        float r  = sigm(gir + hr[v] + br2);
        float z  = sigm(giz + hz[v] + bz2);
        float nn = tanhf(gin + r * (hn[v] + bn2));
        float hnew = (1.0f - z) * nn + z * hp;
        float hout = (1.0f - inv2) * hp + inv2 * hnew;
        h2prev[v] = hout;
        out[((size_t)m * SEQ + t) * (2 * HID) + HID + ncol] = hout;
        h2c[m * HID + ncol] = (_Float16)hout;
      }
    }
    grid_sync(syncv, &localGen);
  }
}

// ---------------- host launch ----------------
extern "C" void kernel_launch(void* const* d_in, const int* in_sizes, int n_in,
                              void* d_out, int out_size, void* d_ws, size_t ws_size,
                              hipStream_t stream) {
  (void)in_sizes; (void)n_in; (void)out_size; (void)ws_size;
  const float* X    = (const float*)d_in[0];
  const float* tau1 = (const float*)d_in[1];
  const float* tau2 = (const float*)d_in[2];
  const float* Wih1 = (const float*)d_in[3];
  const float* Whh1 = (const float*)d_in[4];
  const float* bih1 = (const float*)d_in[5];
  const float* bhh1 = (const float*)d_in[6];
  const float* Wih2 = (const float*)d_in[7];
  const float* Whh2 = (const float*)d_in[8];
  const float* bih2 = (const float*)d_in[9];
  const float* bhh2 = (const float*)d_in[10];
  float* out = (float*)d_out;

  char* ws = (char*)d_ws;
  size_t off = 0;
  auto carve = [&](size_t bytes) -> char* {
    char* p = ws + off;
    off += (bytes + 255) & ~(size_t)255;
    return p;
  };
  _Float16* Xh     = (_Float16*)carve((size_t)ROWS * D_IN * 2);
  _Float16* Wih1s  = (_Float16*)carve((size_t)NT_G3 * KT_IN * 512 * 2);
  _Float16* Wih2xs = (_Float16*)carve((size_t)NT_G3 * KT_IN * 512 * 2);
  _Float16* Whh1s  = (_Float16*)carve((size_t)NT_G3 * KT_H * 512 * 2);
  _Float16* Wih2hs = (_Float16*)carve((size_t)NT_G3 * KT_H * 512 * 2);
  _Float16* Whh2s  = (_Float16*)carve((size_t)NT_G3 * KT_H * 512 * 2);
  float*    GI1    = (float*)carve((size_t)ROWS * G3 * 4);
  float*    GI2    = (float*)carve((size_t)ROWS * G3 * 4);
  _Float16* h1f    = (_Float16*)carve((size_t)2 * BATCH * HID * 2);
  _Float16* h2f    = (_Float16*)carve((size_t)2 * BATCH * HID * 2);
  unsigned* syncv  = (unsigned*)carve(256);

  const int n4 = ROWS * D_IN / 4;
  k_cvt_x<<<(n4 + 255) / 256, 256, 0, stream>>>(X, Xh, n4);
  { int tot = NT_G3 * KT_IN * 512;
    k_pack_w<<<(tot + 255) / 256, 256, 0, stream>>>(Wih1, D_IN, Wih1s);
    k_pack_w<<<(tot + 255) / 256, 256, 0, stream>>>(Wih2, D_IN, Wih2xs); }
  { int tot = NT_G3 * KT_H * 512;
    k_pack_w<<<(tot + 255) / 256, 256, 0, stream>>>(Whh1, HID, Whh1s);
    k_pack_w<<<(tot + 255) / 256, 256, 0, stream>>>(Wih2 + (size_t)D_IN * G3, HID, Wih2hs);
    k_pack_w<<<(tot + 255) / 256, 256, 0, stream>>>(Whh2, HID, Whh2s); }
  k_init<<<64, 256, 0, stream>>>(h1f, h2f, syncv);

  // 2048 mtiles x 12 ngroups = 24576 wave-jobs -> 3072 blocks of 8 waves
  k_gemm_in<<<3072, 256, 0, stream>>>(Xh, Wih1s, Wih2xs, bih1, bih2, GI1, GI2);

  // 144 KB dynamic LDS weight stage per WG (fits 320 KB WGP LDS)
  k_recurrent<<<NWG_REC, THR_REC, LDSW_HALFS * sizeof(_Float16), stream>>>(
      GI1, GI2, Whh1s, Wih2hs, Whh2s, bhh1, bhh2, tau1, tau2, h1f, h2f, out, syncv);
}
